// GraphStructureHead_72146860638579
// MI455X (gfx1250) — matrix operand order repaired and checked by reading, not compiled, not run on hardware
//
#include <hip/hip_runtime.h>
#include <math.h>

typedef __attribute__((ext_vector_type(8)))  float        v8f;
typedef __attribute__((ext_vector_type(8)))  __bf16       v8bf;
typedef __attribute__((ext_vector_type(16))) __bf16       v16bf;
typedef __attribute__((ext_vector_type(4)))  unsigned int v4u;
typedef __attribute__((ext_vector_type(8)))  int          v8i;
typedef __attribute__((ext_vector_type(4)))  int          v4i;

#define B_      16
#define C_      256
#define HW_     128
#define NROOM   10
#define ASTR    40   // LDS A row stride (bf16); 80B rows keep 16B alignment

// ---------------- workspace layout (bytes) ----------------
// Wb1  : 72*128*32 bf16 = 589824          @ 0
// Wb2  : 36*16*32  bf16 = 36864           @ 589824
// h    : 16*128*128*128 bf16 = 67108864   @ 626688
// flat : 16*16384 f32 = 1048576           @ 67735552
// t1   : 16*256  f32  = 16384             @ 68784128

// ---- TDM: 1-D contiguous tile (nbytes, 8B elements) global -> LDS ----
__device__ __forceinline__ void tdm_load_1d(unsigned lds_off, const void* gptr,
                                            unsigned nbytes) {
    unsigned long long ga = (unsigned long long)(uintptr_t)gptr;
    unsigned n8 = nbytes >> 3;                 // element count (8B each)
    v4u g0;
    g0[0] = 1u;                                           // count=1, no gather
    g0[1] = lds_off;                                      // lds_addr (bytes)
    g0[2] = (unsigned)(ga & 0xffffffffu);                 // global_addr[31:0]
    g0[3] = (unsigned)((ga >> 32) & 0x1ffffffu) | (2u << 30); // addr[56:32] | type=2
    v8i g1;
    g1[0] = (int)(3u << 16);                              // data_size=3 (8 bytes)
    g1[1] = (int)((n8 & 0xffffu) << 16);                  // tensor_dim0[15:0]
    g1[2] = (int)((n8 >> 16) & 0xffffu) | (1 << 16);      // tensor_dim0[31:16], tensor_dim1=1
    g1[3] = (int)((n8 & 0xffffu) << 16);                  // tile_dim0
    g1[4] = 1;                                            // tile_dim1=1
    g1[5] = (int)n8;                                      // tensor_dim0_stride lo
    g1[6] = 0;
    g1[7] = 0;
    v4i z4 = {0, 0, 0, 0};
    v8i z8 = {0, 0, 0, 0, 0, 0, 0, 0};
    __builtin_amdgcn_tensor_load_to_lds(g0, g1, z4, z4, z8, 0);
}

// ---- A fragment (16x32 bf16 WMMA layout) from unshifted LDS row ms ----
__device__ __forceinline__ v16bf frag_a(const __bf16* Alds, int ms, int hl) {
    v16bf a = {};
    if (ms >= 0 && ms < HW_) {
        const __bf16* ap = Alds + ms * ASTR + hl * 16;
        v8bf lo = *(const v8bf*)ap;
        v8bf hi = *(const v8bf*)(ap + 8);
        a = __builtin_shufflevector(lo, hi, 0,1,2,3,4,5,6,7,8,9,10,11,12,13,14,15);
    }
    return a;
}

// ============ weight repack: conv1_w [128][256][3][3] -> Wb1[chunk][n][kk] bf16
// chunk = (ky*8+cb)*3 + kx  (kx innermost so A-staging happens once per 3 chunks)
__global__ void prep_w1(const float* __restrict__ w, __bf16* __restrict__ Wb1) {
    int idx = blockIdx.x * 256 + threadIdx.x;           // 294912
    if (idx >= 72 * 128 * 32) return;
    int kk = idx & 31;
    int n  = (idx >> 5) & 127;
    int chunk = idx >> 12;
    int kx = chunk % 3, t = chunk / 3;
    int cb = t & 7, ky = t >> 3;
    int ci = cb * 32 + kk;
    Wb1[idx] = (__bf16)w[((size_t)n * 256 + ci) * 9 + ky * 3 + kx];
}

// conv2_w [10][128][3][3] -> Wb2[chunk][n][kk], chunk=(ky*4+cb)*3+kx, n padded to 16
__global__ void prep_w2(const float* __restrict__ w, __bf16* __restrict__ Wb2) {
    int idx = blockIdx.x * 256 + threadIdx.x;           // 18432
    if (idx >= 36 * 16 * 32) return;
    int kk = idx & 31;
    int n  = (idx >> 5) & 15;
    int chunk = idx >> 9;
    int kx = chunk % 3, t = chunk / 3;
    int cb = t & 3, ky = t >> 2;
    int ci = cb * 32 + kk;
    float v = (n < NROOM) ? w[((size_t)n * 128 + ci) * 9 + ky * 3 + kx] : 0.f;
    Wb2[idx] = (__bf16)v;
}

// ============ conv1: h = relu(conv3x3(features)+b) bf16 ; grid 2048, block 256
__global__ __launch_bounds__(256) void conv1_wmma(
    const float* __restrict__ feat, const __bf16* __restrict__ Wb1,
    const float* __restrict__ bias, __bf16* __restrict__ hout)
{
    __shared__ __attribute__((aligned(32))) __bf16 Alds[HW_ * ASTR];
    __shared__ __attribute__((aligned(32))) __bf16 Blds[2][128 * 32];
    const int tid  = threadIdx.x;
    const int wave = tid >> 5;
    const int lane = tid & 31;
    const int hl   = lane >> 4;
    const int l15  = lane & 15;
    const int b = blockIdx.x >> 7;
    const int y = blockIdx.x & 127;
    const int kkT = tid >> 3;            // staging channel 0..31
    const int x0  = (tid & 7) << 4;      // staging 16 consecutive x

    v8f acc[8] = {};

    if (tid < 32)  // prefetch B tile for chunk 0 via TDM
        tdm_load_1d((unsigned)(uintptr_t)&Blds[0][0], Wb1, 8192);

    for (int c = 0; c < 72; ++c) {
        const int kx = c % 3;
        const int t3 = c / 3;
        const int cb = t3 & 7;
        const int ky = t3 >> 3;
        const int yy = y + ky - 1;
        const int sx = kx - 1;

        __syncthreads();   // prior iter done reading Alds / Blds[(c+1)&1]

        if (kx == 0) {     // stage unshifted input row for this (ky, cb)
            int ci = cb * 32 + kkT;
            __bf16* dst = &Alds[x0 * ASTR + kkT];
            if (yy < 0 || yy >= HW_) {
                #pragma unroll
                for (int i = 0; i < 16; ++i) dst[i * ASTR] = (__bf16)0.f;
            } else {
                const float4* s4 =
                    (const float4*)(feat + (((size_t)b * C_ + ci) * HW_ + yy) * HW_ + x0);
                #pragma unroll
                for (int q = 0; q < 4; ++q) {
                    float4 v = s4[q];
                    dst[(q * 4 + 0) * ASTR] = (__bf16)v.x;
                    dst[(q * 4 + 1) * ASTR] = (__bf16)v.y;
                    dst[(q * 4 + 2) * ASTR] = (__bf16)v.z;
                    dst[(q * 4 + 3) * ASTR] = (__bf16)v.w;
                }
                if (cb < 7)  // prefetch next channel block's row
                    __builtin_prefetch((const float*)s4 + 32 * HW_ * HW_, 0, 1);
            }
        }
        if (tid < 32 && c + 1 < 72)   // async B tile for next chunk
            tdm_load_1d((unsigned)(uintptr_t)&Blds[(c + 1) & 1][0],
                        Wb1 + (size_t)(c + 1) * 4096, 8192);
        if (c + 1 < 72) __builtin_amdgcn_s_wait_tensorcnt(1);
        else            __builtin_amdgcn_s_wait_tensorcnt(0);
        __syncthreads();   // A staged + B tile c landed

        const __bf16* Bl = &Blds[c & 1][0];
        v16bf bfr[8];
        #pragma unroll
        for (int ns = 0; ns < 8; ++ns)
            bfr[ns] = *(const v16bf*)(Bl + (ns * 16 + l15) * 32 + hl * 16);
        v16bf afrag = frag_a(Alds, wave * 16 + l15 + sx, hl);
        #pragma unroll
        for (int ns = 0; ns < 8; ++ns)
            acc[ns] = __builtin_amdgcn_wmma_f32_16x16x32_bf16(
                false, afrag, false, bfr[ns], (short)0, acc[ns], false, false);
    }

    // epilogue: each lane owns 8 consecutive x for one channel -> one 16B store
    const int xo = wave * 16 + hl * 8;
    #pragma unroll
    for (int ns = 0; ns < 8; ++ns) {
        int n = ns * 16 + l15;
        float bs = bias[n];
        v8bf o;
        #pragma unroll
        for (int r = 0; r < 8; ++r) {
            float v = acc[ns][r] + bs;
            o[r] = (__bf16)(v > 0.f ? v : 0.f);
        }
        *(v8bf*)(hout + (((size_t)b * 128 + n) * HW_ + y) * HW_ + xo) = o;
    }
}

// ============ conv2: room = sigmoid(conv3x3(h)+b) f32 ; grid 2048, N=16(10)
__global__ __launch_bounds__(256) void conv2_wmma(
    const __bf16* __restrict__ hin, const __bf16* __restrict__ Wb2,
    const float* __restrict__ bias, float* __restrict__ room)
{
    __shared__ __attribute__((aligned(32))) __bf16 Alds[HW_ * ASTR];
    __shared__ __attribute__((aligned(32))) __bf16 Blds[2][16 * 32];
    const int tid  = threadIdx.x;
    const int wave = tid >> 5;
    const int lane = tid & 31;
    const int hl   = lane >> 4;
    const int l15  = lane & 15;
    const int b = blockIdx.x >> 7;
    const int y = blockIdx.x & 127;
    const int kkT = tid >> 3;
    const int x0  = (tid & 7) << 4;

    v8f acc = {};

    if (tid < 32)
        tdm_load_1d((unsigned)(uintptr_t)&Blds[0][0], Wb2, 1024);

    for (int c = 0; c < 36; ++c) {
        const int kx = c % 3;
        const int t3 = c / 3;
        const int cb = t3 & 3;
        const int ky = t3 >> 2;
        const int yy = y + ky - 1;
        const int sx = kx - 1;

        __syncthreads();

        if (kx == 0) {
            int ci = cb * 32 + kkT;
            __bf16* dst = &Alds[x0 * ASTR + kkT];
            if (yy < 0 || yy >= HW_) {
                #pragma unroll
                for (int i = 0; i < 16; ++i) dst[i * ASTR] = (__bf16)0.f;
            } else {
                const v8bf* s8 =
                    (const v8bf*)(hin + (((size_t)b * 128 + ci) * HW_ + yy) * HW_ + x0);
                v8bf a0 = s8[0], a1 = s8[1];
                #pragma unroll
                for (int i = 0; i < 8; ++i) {
                    dst[i * ASTR]       = a0[i];
                    dst[(i + 8) * ASTR] = a1[i];
                }
            }
        }
        if (tid < 32 && c + 1 < 36)
            tdm_load_1d((unsigned)(uintptr_t)&Blds[(c + 1) & 1][0],
                        Wb2 + (size_t)(c + 1) * 512, 1024);
        if (c + 1 < 36) __builtin_amdgcn_s_wait_tensorcnt(1);
        else            __builtin_amdgcn_s_wait_tensorcnt(0);
        __syncthreads();

        v16bf bfrag = *(const v16bf*)(&Blds[c & 1][0] + l15 * 32 + hl * 16);
        v16bf afrag = frag_a(Alds, wave * 16 + l15 + sx, hl);
        acc = __builtin_amdgcn_wmma_f32_16x16x32_bf16(
            false, afrag, false, bfrag, (short)0, acc, false, false);
    }

    const int n = l15;
    if (n < NROOM) {
        float bs = bias[n];
        float* dst = room + (((size_t)b * NROOM + n) * HW_ + y) * HW_ + wave * 16 + hl * 8;
        float4 o0, o1;
        float t;
        t = acc[0] + bs; o0.x = 1.f / (1.f + __expf(-t));
        t = acc[1] + bs; o0.y = 1.f / (1.f + __expf(-t));
        t = acc[2] + bs; o0.z = 1.f / (1.f + __expf(-t));
        t = acc[3] + bs; o0.w = 1.f / (1.f + __expf(-t));
        t = acc[4] + bs; o1.x = 1.f / (1.f + __expf(-t));
        t = acc[5] + bs; o1.y = 1.f / (1.f + __expf(-t));
        t = acc[6] + bs; o1.z = 1.f / (1.f + __expf(-t));
        t = acc[7] + bs; o1.w = 1.f / (1.f + __expf(-t));
        ((float4*)dst)[0] = o0;
        ((float4*)dst)[1] = o1;
    }
}

// ============ adaptive avg pool 128x128 -> 8x8 ; flat[b][c*64 + py*8 + px]
__global__ void pool_kernel(const float* __restrict__ feat, float* __restrict__ flat) {
    int b = blockIdx.x >> 8;
    int c = blockIdx.x & 255;
    int py = threadIdx.x >> 3, px = threadIdx.x & 7;
    const float* base = feat + (((size_t)b * C_ + c) * HW_ + py * 16) * HW_ + px * 16;
    float s = 0.f;
    for (int r = 0; r < 16; ++r) {
        const float4* rp = (const float4*)(base + r * HW_);
        #pragma unroll
        for (int q = 0; q < 4; ++q) {
            float4 v = rp[q];
            s += v.x + v.y + v.z + v.w;
        }
    }
    flat[(size_t)b * 16384 + c * 64 + threadIdx.x] = s * (1.f / 256.f);
}

// ============ lin1: t1[b][j] = relu(flat[b] . lin1_w[j] + b1[j])
__global__ __launch_bounds__(256) void lin1_kernel(
    const float* __restrict__ flat, const float* __restrict__ w,
    const float* __restrict__ bv, float* __restrict__ t1)
{
    int j = blockIdx.x;
    const float* wr = w + (size_t)j * 16384;
    float part[16];
    #pragma unroll
    for (int b = 0; b < 16; ++b) part[b] = 0.f;
    for (int i = 0; i < 64; ++i) {
        int k = threadIdx.x + (i << 8);
        float wv = wr[k];
        #pragma unroll
        for (int b = 0; b < 16; ++b) part[b] += wv * flat[(size_t)b * 16384 + k];
    }
    __shared__ float red[256];
    for (int b = 0; b < 16; ++b) {
        red[threadIdx.x] = part[b];
        __syncthreads();
        for (int s = 128; s > 0; s >>= 1) {
            if (threadIdx.x < s) red[threadIdx.x] += red[threadIdx.x + s];
            __syncthreads();
        }
        if (threadIdx.x == 0) t1[b * 256 + j] = fmaxf(red[0] + bv[j], 0.f);
        __syncthreads();
    }
}

// ============ lin2: feat_out[b][j] = t1[b] . lin2_w[j] + b2[j]
__global__ void lin2_kernel(const float* __restrict__ t1, const float* __restrict__ w,
                            const float* __restrict__ bv, float* __restrict__ feat_out)
{
    int j = threadIdx.x;
    for (int b = 0; b < 16; ++b) {
        float s = bv[j];
        const float* wr = w + (size_t)j * 256;
        for (int k = 0; k < 256; ++k) s += t1[b * 256 + k] * wr[k];
        feat_out[b * 128 + j] = s;
    }
}

// ============ adjacency MLP (1 block x 256)
__global__ void adj_kernel(const float* __restrict__ feat_out,
                           const float* __restrict__ w1, const float* __restrict__ b1,
                           const float* __restrict__ w2, const float* __restrict__ b2,
                           const float* __restrict__ w3, const float* __restrict__ b3,
                           float* __restrict__ adj)
{
    __shared__ float sfeat[2048];
    __shared__ float a1[64];
    __shared__ float a2[32];
    __shared__ float pv;
    int tid = threadIdx.x;
    for (int i = tid; i < 2048; i += 256) sfeat[i] = feat_out[i];
    __syncthreads();
    for (int b = 0; b < 16; ++b) {
        if (tid < 64) {
            float s = b1[tid];
            const float* wr = w1 + tid * 256;
            const float* f = sfeat + b * 128;
            for (int k = 0; k < 256; ++k) s += wr[k] * f[k & 127];
            a1[tid] = fmaxf(s, 0.f);
        }
        __syncthreads();
        if (tid < 32) {
            float s = b2[tid];
            const float* wr = w2 + tid * 64;
            for (int k = 0; k < 64; ++k) s += wr[k] * a1[k];
            a2[tid] = fmaxf(s, 0.f);
        }
        __syncthreads();
        if (tid == 0) {
            float s = b3[0];
            for (int k = 0; k < 32; ++k) s += w3[k] * a2[k];
            pv = 1.f / (1.f + __expf(-s));
        }
        __syncthreads();
        if (tid < 100) {
            int i = tid / 10, j = tid % 10;
            adj[b * 100 + tid] = (i == j) ? 0.f : pv;
        }
        __syncthreads();
    }
}

extern "C" void kernel_launch(void* const* d_in, const int* in_sizes, int n_in,
                              void* d_out, int out_size, void* d_ws, size_t ws_size,
                              hipStream_t stream) {
    (void)in_sizes; (void)n_in; (void)out_size; (void)ws_size;
    const float* features = (const float*)d_in[0];
    const float* conv1_w  = (const float*)d_in[1];
    const float* conv1_b  = (const float*)d_in[2];
    const float* conv2_w  = (const float*)d_in[3];
    const float* conv2_b  = (const float*)d_in[4];
    const float* lin1_w   = (const float*)d_in[5];
    const float* lin1_b   = (const float*)d_in[6];
    const float* lin2_w   = (const float*)d_in[7];
    const float* lin2_b   = (const float*)d_in[8];
    const float* adj1_w   = (const float*)d_in[9];
    const float* adj1_b   = (const float*)d_in[10];
    const float* adj2_w   = (const float*)d_in[11];
    const float* adj2_b   = (const float*)d_in[12];
    const float* adj3_w   = (const float*)d_in[13];
    const float* adj3_b   = (const float*)d_in[14];

    char* ws = (char*)d_ws;
    __bf16* Wb1 = (__bf16*)(ws + 0);
    __bf16* Wb2 = (__bf16*)(ws + 589824);
    __bf16* hbf = (__bf16*)(ws + 626688);
    float*  flat = (float*)(ws + 67735552);
    float*  t1   = (float*)(ws + 68784128);

    float* out      = (float*)d_out;
    float* room     = out;                 // 2621440
    float* feat_out = out + 2621440;       // 2048
    float* adj      = out + 2623488;       // 1600

    prep_w1<<<1152, 256, 0, stream>>>(conv1_w, Wb1);
    prep_w2<<<72,   256, 0, stream>>>(conv2_w, Wb2);
    conv1_wmma<<<2048, 256, 0, stream>>>(features, Wb1, conv1_b, hbf);
    conv2_wmma<<<2048, 256, 0, stream>>>(hbf, Wb2, conv2_b, room);
    pool_kernel<<<4096, 64, 0, stream>>>(features, flat);
    lin1_kernel<<<256, 256, 0, stream>>>(flat, lin1_w, lin1_b, t1);
    lin2_kernel<<<1, 128, 0, stream>>>(t1, lin2_w, lin2_b, feat_out);
    adj_kernel<<<1, 256, 0, stream>>>(feat_out, adj1_w, adj1_b, adj2_w, adj2_b,
                                      adj3_w, adj3_b, adj);
}
